// GT_12008728560087
// MI455X (gfx1250) — compile-verified, matrix operand order checked by reading, not run on hardware
//
#include <hip/hip_runtime.h>

#define NN 50000
#define EE 800000
#define G_ 64

typedef __attribute__((ext_vector_type(8)))  float v8f;
typedef __attribute__((ext_vector_type(16))) __bf16 v16bf;
typedef __attribute__((ext_vector_type(8)))  unsigned short u16x8;

struct Frag32 { u16x8 lo, hi; };

__device__ __forceinline__ unsigned short f2bf(float x) {
    unsigned u = __float_as_uint(x);
    u += 0x7FFFu + ((u >> 16) & 1u);           // round-to-nearest-even
    return (unsigned short)(u >> 16);
}
// ordered-uint encoding so unsigned atomicMax == float max; 0 encodes below -inf
__device__ __forceinline__ unsigned encf(float v) {
    unsigned b = __float_as_uint(v);
    return (b & 0x80000000u) ? ~b : (b | 0x80000000u);
}
__device__ __forceinline__ float decf(unsigned e) {
    return (e & 0x80000000u) ? __uint_as_float(e & 0x7FFFFFFFu) : __uint_as_float(~e);
}

// 16B async DMA global -> LDS (ASYNCcnt path, CDNA5 ISA)
__device__ __forceinline__ void async_cp16(unsigned lds_off, const void* gptr) {
    asm volatile("global_load_async_to_lds_b128 %0, %1, off"
                 :: "v"(lds_off), "v"((unsigned long long)(size_t)gptr)
                 : "memory");
}
__device__ __forceinline__ void wait_async0() {
    asm volatile("s_wait_asynccnt 0x0" ::: "memory");
}
__device__ __forceinline__ unsigned lds_off32(const void* p) {
    // generic LDS pointer = {SHARED_BASE, offset32}; low 32 bits are the LDS byte offset
    return (unsigned)(size_t)p;
}

// ---------------- weight packing (f32 -> bf16) ----------------
__global__ __launch_bounds__(256)
void pack_weights(const float* __restrict__ Wq, const float* __restrict__ Wk,
                  const float* __restrict__ Wv, const float* __restrict__ Ws,
                  const float* __restrict__ bq, const float* __restrict__ bk,
                  const float* __restrict__ bv, const float* __restrict__ bs,
                  const float* __restrict__ g1w,
                  unsigned short* __restrict__ wall, float* __restrict__ ball,
                  unsigned short* __restrict__ g1bf)
{
    int idx = blockIdx.x * 256 + threadIdx.x;
    if (idx < 512 * 128) {
        int o = idx >> 7, k = idx & 127;
        float v;
        if      (o < 128) v = Wq[o * 128 + k];
        else if (o < 256) v = Wk[(o - 128) * 128 + k];
        else if (o < 384) v = Wv[(o - 256) * 128 + k];
        else              v = Ws[(o - 384) * 128 + k];
        wall[idx] = f2bf(v);
    }
    if (idx < 16384) g1bf[idx] = f2bf(g1w[idx]);
    if (idx < 512) {
        float b;
        if      (idx < 128) b = bq[idx];
        else if (idx < 256) b = bk[idx - 128];
        else if (idx < 384) b = bv[idx - 256];
        else                b = bs[idx - 384];
        ball[idx] = b;
    }
}

// ---------------- input projection: h = [x|pe] @ W.T + b ----------------
__global__ __launch_bounds__(128)
void node_in_kernel(const float* __restrict__ x, const float* __restrict__ pe,
                    const float* __restrict__ W, const float* __restrict__ b,
                    float* __restrict__ h, unsigned short* __restrict__ hbf)
{
    __shared__ float sin_[47];
    int n = blockIdx.x, t = threadIdx.x;
    if (t < 32)      sin_[t] = x[(size_t)n * 32 + t];
    else if (t < 47) sin_[t] = pe[(size_t)n * 15 + (t - 32)];
    __syncthreads();
    float acc = b[t];
    const float* wr = W + (size_t)t * 47;
    for (int k = 0; k < 47; ++k) acc += sin_[k] * wr[k];
    h[(size_t)n * 128 + t]   = acc;
    hbf[(size_t)n * 128 + t] = f2bf(acc);
}

// ---- WMMA bf16 GEMM: Out[M,NOUT] = A[M,128] @ W[NOUT,128]^T + bias ----
// A and W are bf16 in global; both tiles staged to LDS via async DMA.
__global__ __launch_bounds__(256)
void gemm_bf16_kernel(const unsigned short* __restrict__ Abf,
                      const unsigned short* __restrict__ W,
                      const float* __restrict__ bias, float* __restrict__ Out,
                      int M, int NOUT, int do_relu)
{
    __shared__ unsigned short As[64][136];     // padded stride: bank-conflict-free
    __shared__ unsigned short Bs[128][136];
    const int tid    = threadIdx.x;
    const int m0     = blockIdx.x * 64;
    const int nbase0 = blockIdx.y * 128;

    { // A tile: 64 rows x 256B, 4 threads/row x 4 chunks of 16B
        int r = tid >> 2, seg = tid & 3;
        int node = m0 + r; if (node > M - 1) node = M - 1;   // clamp: no divergence
        unsigned ldsRow = lds_off32(&As[r][0]);
        const char* g = (const char*)(Abf + (size_t)node * 128);
        for (int j = 0; j < 4; ++j) {
            int ci = seg + j * 4;
            async_cp16(ldsRow + ci * 16, g + ci * 16);
        }
    }
    { // B tile: 128 rows x 256B, 2 threads/row x 8 chunks of 16B
        int r = tid >> 1, seg = tid & 1;
        unsigned ldsRow = lds_off32(&Bs[r][0]);
        const char* g = (const char*)(W + (size_t)(nbase0 + r) * 128);
        for (int j = 0; j < 8; ++j) {
            int ci = seg * 8 + j;
            async_cp16(ldsRow + ci * 16, g + ci * 16);
        }
    }
    wait_async0();
    __syncthreads();

    const int w       = tid >> 5;
    const int lane    = tid & 31;
    const int waveM   = w >> 1;                     // 0..3  (4 x 16 rows)
    const int waveN   = w & 1;                      // 0..1  (2 x 64 cols)
    const int colLane = lane & 15;
    const int khalf   = (lane >> 4) << 3;

    v8f acc[4];
    for (int nt = 0; nt < 4; ++nt) {
        float b = bias[nbase0 + waveN * 64 + nt * 16 + colLane];
        for (int r = 0; r < 8; ++r) acc[nt][r] = b;
    }

    for (int kk = 0; kk < 4; ++kk) {
        int kb = kk * 32 + khalf;
        Frag32 fa;
        const unsigned short* ap = &As[waveM * 16 + colLane][kb];
        fa.lo = *(const u16x8*)(ap);
        fa.hi = *(const u16x8*)(ap + 16);
        v16bf afrag = __builtin_bit_cast(v16bf, fa);
        for (int nt = 0; nt < 4; ++nt) {
            const unsigned short* bp = &Bs[waveN * 64 + nt * 16 + colLane][kb];
            Frag32 fb;
            fb.lo = *(const u16x8*)(bp);
            fb.hi = *(const u16x8*)(bp + 16);
            v16bf bfrag = __builtin_bit_cast(v16bf, fb);
            acc[nt] = __builtin_amdgcn_wmma_f32_16x16x32_bf16(
                false, afrag, false, bfrag, (short)0, acc[nt], false, false);
        }
    }

    int mrow0 = m0 + waveM * 16 + ((lane >> 4) << 3);
    for (int nt = 0; nt < 4; ++nt) {
        int col = nbase0 + waveN * 64 + nt * 16 + colLane;
        for (int r = 0; r < 8; ++r) {
            int node = mrow0 + r;
            if (node < M) {
                float v = acc[nt][r];
                if (do_relu) v = fmaxf(v, 0.f);
                Out[(size_t)node * NOUT + col] = v;
            }
        }
    }
}

// ---------------- edge attention logits + segment max ----------------
__global__ __launch_bounds__(256)
void edge_alpha_kernel(const float* __restrict__ qkvs, const int* __restrict__ src,
                       const int* __restrict__ dst, float* __restrict__ exb,
                       unsigned* __restrict__ amax)
{
    int idx = blockIdx.x * 256 + threadIdx.x;       // EE*4 = 3.2M
    if (idx >= EE * 4) return;
    int e = idx >> 2, hd = idx & 3;
    int s = src[e], d = dst[e];
    const float4* qp = (const float4*)(qkvs + (size_t)d * 512 + hd * 32);
    const float4* kp = (const float4*)(qkvs + (size_t)s * 512 + 128 + hd * 32);
    float acc = 0.f;
    for (int j = 0; j < 8; ++j) {
        float4 a = qp[j], b = kp[j];
        acc += a.x * b.x + a.y * b.y + a.z * b.z + a.w * b.w;
    }
    float alpha = acc * 0.17677669529663687f;       // 1/sqrt(32)
    exb[idx] = alpha;
    atomicMax(&amax[(size_t)d * 4 + hd], encf(alpha));
}

// ---------------- exp + segment denom ----------------
__global__ __launch_bounds__(256)
void edge_exp_kernel(float* __restrict__ exb, const int* __restrict__ dst,
                     const unsigned* __restrict__ amax, float* __restrict__ denom)
{
    int idx = blockIdx.x * 256 + threadIdx.x;
    if (idx >= EE * 4) return;
    int e = idx >> 2, hd = idx & 3;
    int d = dst[e];
    float ex = __expf(exb[idx] - decf(amax[(size_t)d * 4 + hd]));
    exb[idx] = ex;
    atomicAdd(&denom[(size_t)d * 4 + hd], ex);
}

// ---------------- weighted message scatter-add ----------------
__global__ __launch_bounds__(256)
void edge_scatter_kernel(const float* __restrict__ qkvs, const int* __restrict__ src,
                         const int* __restrict__ dst, const float* __restrict__ exb,
                         const float* __restrict__ denom, float* __restrict__ outagg)
{
    int idx = blockIdx.x * 256 + threadIdx.x;       // EE*128 = 102.4M exact
    int e = idx >> 7, c = idx & 127, hd = c >> 5;
    int s = src[e], d = dst[e];
    float w = exb[(size_t)e * 4 + hd] / denom[(size_t)d * 4 + hd];
    atomicAdd(&outagg[(size_t)d * 128 + c], w * qkvs[(size_t)s * 512 + 256 + c]);
}

// ---------------- beta-gated residual update (one wave per node) ----------------
__global__ __launch_bounds__(256)
void node_update_kernel(float* __restrict__ h, unsigned short* __restrict__ hbf,
                        const float* __restrict__ qkvs,
                        const float* __restrict__ outagg, const float* __restrict__ wbeta)
{
    int n = blockIdx.x * 8 + (threadIdx.x >> 5);
    int lane = threadIdx.x & 31;
    if (n >= NN) return;
    int c = lane * 4;
    float4 o  = *(const float4*)(outagg + (size_t)n * 128 + c);
    float4 xr = *(const float4*)(qkvs + (size_t)n * 512 + 384 + c);
    float4 hv = *(const float4*)(h + (size_t)n * 128 + c);
    float4 w1 = *(const float4*)(wbeta + c);
    float4 w2 = *(const float4*)(wbeta + 128 + c);
    float4 w3 = *(const float4*)(wbeta + 256 + c);
    float p = o.x * w1.x + o.y * w1.y + o.z * w1.z + o.w * w1.w
            + xr.x * w2.x + xr.y * w2.y + xr.z * w2.z + xr.w * w2.w
            + (o.x - xr.x) * w3.x + (o.y - xr.y) * w3.y
            + (o.z - xr.z) * w3.z + (o.w - xr.w) * w3.w;
    for (int m = 16; m > 0; m >>= 1) p += __shfl_xor(p, m, 32);
    float beta = 1.f / (1.f + __expf(-p));
    float4 r;
    r.x = fmaxf(hv.x + beta * xr.x + (1.f - beta) * o.x, 0.f);
    r.y = fmaxf(hv.y + beta * xr.y + (1.f - beta) * o.y, 0.f);
    r.z = fmaxf(hv.z + beta * xr.z + (1.f - beta) * o.z, 0.f);
    r.w = fmaxf(hv.w + beta * xr.w + (1.f - beta) * o.w, 0.f);
    *(float4*)(h + (size_t)n * 128 + c) = r;
    ushort4 hb;
    hb.x = f2bf(r.x); hb.y = f2bf(r.y); hb.z = f2bf(r.z); hb.w = f2bf(r.w);
    *(ushort4*)(hbf + (size_t)n * 128 + c) = hb;
}

// ---------------- gate scalar + graph max (one wave per node) ----------------
__global__ __launch_bounds__(256)
void gate_kernel(const float* __restrict__ t, const float* __restrict__ g2w,
                 const float* __restrict__ g2b, const int* __restrict__ batch,
                 float* __restrict__ gate, unsigned* __restrict__ gmax)
{
    int n = blockIdx.x * 8 + (threadIdx.x >> 5);
    int lane = threadIdx.x & 31;
    if (n >= NN) return;
    int c = lane * 4;
    float4 tv = *(const float4*)(t + (size_t)n * 128 + c);
    float4 wv = *(const float4*)(g2w + c);
    float p = tv.x * wv.x + tv.y * wv.y + tv.z * wv.z + tv.w * wv.w;
    for (int m = 16; m > 0; m >>= 1) p += __shfl_xor(p, m, 32);
    if (lane == 0) {
        float g = p + g2b[0];
        gate[n] = g;
        atomicMax(&gmax[batch[n]], encf(g));
    }
}

__global__ __launch_bounds__(256)
void gexp_kernel(const float* __restrict__ gate, const int* __restrict__ batch,
                 const unsigned* __restrict__ gmax, float* __restrict__ gex,
                 float* __restrict__ gden)
{
    int n = blockIdx.x * 256 + threadIdx.x;
    if (n >= NN) return;
    int b = batch[n];
    float e = __expf(gate[n] - decf(gmax[b]));
    gex[n] = e;
    atomicAdd(&gden[b], e);
}

__global__ __launch_bounds__(256)
void gsum_kernel(const float* __restrict__ h, const float* __restrict__ gex,
                 const float* __restrict__ gden, const int* __restrict__ batch,
                 float* __restrict__ gsum)
{
    int idx = blockIdx.x * 256 + threadIdx.x;       // NN*128 = 6.4M exact
    int n = idx >> 7, c = idx & 127;
    int b = batch[n];
    float w = gex[n] / gden[b];
    atomicAdd(&gsum[(size_t)b * 128 + c], w * h[(size_t)n * 128 + c]);
}

// ---------------- readout MLP (one block per graph) ----------------
__global__ __launch_bounds__(128)
void readout_kernel(const float* __restrict__ g, const float* __restrict__ r1w,
                    const float* __restrict__ r1b, const float* __restrict__ r2w,
                    const float* __restrict__ r2b, float* __restrict__ out)
{
    __shared__ float red[128];
    int gid = blockIdx.x, t = threadIdx.x;
    const float* gr = g + (size_t)gid * 128;
    float acc = r1b[t];
    const float* wr = r1w + (size_t)t * 128;
    for (int k = 0; k < 128; ++k) acc += gr[k] * wr[k];
    acc = fmaxf(acc, 0.f);
    red[t] = acc * r2w[t];
    __syncthreads();
    for (int s = 64; s > 0; s >>= 1) {
        if (t < s) red[t] += red[t + s];
        __syncthreads();
    }
    if (t == 0) out[gid] = red[0] + r2b[0];
}

extern "C" void kernel_launch(void* const* d_in, const int* in_sizes, int n_in,
                              void* d_out, int out_size, void* d_ws, size_t ws_size,
                              hipStream_t stream)
{
    const float* x     = (const float*)d_in[0];
    const int*   ei    = (const int*)d_in[1];
    const int*   batch = (const int*)d_in[2];
    const float* pe    = (const float*)d_in[3];
    const float* niw   = (const float*)d_in[4];
    const float* nib   = (const float*)d_in[5];
    const float* Wq = (const float*)d_in[6];  const float* bq = (const float*)d_in[7];
    const float* Wk = (const float*)d_in[8];  const float* bk = (const float*)d_in[9];
    const float* Wv = (const float*)d_in[10]; const float* bv = (const float*)d_in[11];
    const float* Ws = (const float*)d_in[12]; const float* bs = (const float*)d_in[13];
    const float* Wbeta = (const float*)d_in[14];
    const float* g1w = (const float*)d_in[15]; const float* g1b = (const float*)d_in[16];
    const float* g2w = (const float*)d_in[17]; const float* g2b = (const float*)d_in[18];
    const float* r1w = (const float*)d_in[19]; const float* r1b = (const float*)d_in[20];
    const float* r2w = (const float*)d_in[21]; const float* r2b = (const float*)d_in[22];

    const int* srcI = ei;
    const int* dstI = ei + EE;

    char* ws = (char*)d_ws;
    size_t off = 0;
    auto take = [&](size_t bytes) -> char* {
        char* p = ws + off;
        off += (bytes + 255) & ~(size_t)255;
        return p;
    };
    unsigned short* wall = (unsigned short*)take(512 * 128 * 2);
    unsigned short* g1bf = (unsigned short*)take(128 * 128 * 2);
    float* ball   = (float*)take(512 * 4);
    float* h      = (float*)take((size_t)NN * 128 * 4);
    unsigned short* hbf = (unsigned short*)take((size_t)NN * 128 * 2);
    float* qkvs   = (float*)take((size_t)NN * 512 * 4);
    float* outagg = (float*)take((size_t)NN * 128 * 4);   // |--- contiguous region
    unsigned* amax = (unsigned*)take((size_t)NN * 4 * 4); //     zeroed together
    float* denom  = (float*)take((size_t)NN * 4 * 4);     // ---|
    float* exb    = (float*)take((size_t)EE * 4 * 4);
    float* gate   = (float*)take((size_t)NN * 4);
    float* gex    = (float*)take((size_t)NN * 4);
    unsigned* gmax = (unsigned*)take(G_ * 4);             // |--- contiguous region
    float* gden   = (float*)take(G_ * 4);                 //     zeroed together
    float* gsum   = (float*)take((size_t)G_ * 128 * 4);   // ---|

    const int gemmGridX = (NN + 63) / 64;                 // 782

    pack_weights<<<256, 256, 0, stream>>>(Wq, Wk, Wv, Ws, bq, bk, bv, bs, g1w,
                                          wall, ball, g1bf);
    node_in_kernel<<<NN, 128, 0, stream>>>(x, pe, niw, nib, h, hbf);

    const size_t layerZero = (size_t)NN * 128 * 4 + 2 * (size_t)NN * 4 * 4; // outagg+amax+denom
    for (int layer = 0; layer < 4; ++layer) {
        hipMemsetAsync(outagg, 0, layerZero, stream);
        gemm_bf16_kernel<<<dim3(gemmGridX, 4), 256, 0, stream>>>(hbf, wall, ball, qkvs,
                                                                 NN, 512, 0);
        edge_alpha_kernel<<<(EE * 4) / 256, 256, 0, stream>>>(qkvs, srcI, dstI, exb, amax);
        edge_exp_kernel<<<(EE * 4) / 256, 256, 0, stream>>>(exb, dstI, amax, denom);
        edge_scatter_kernel<<<(EE / 2), 256, 0, stream>>>(qkvs, srcI, dstI, exb, denom, outagg);
        node_update_kernel<<<(NN + 7) / 8, 256, 0, stream>>>(h, hbf, qkvs, outagg, Wbeta);
    }

    hipMemsetAsync(gmax, 0, 256 + 256 + (size_t)G_ * 128 * 4, stream);
    gemm_bf16_kernel<<<dim3(gemmGridX, 1), 256, 0, stream>>>(hbf, g1bf, g1b, qkvs, NN, 128, 1);
    gate_kernel<<<(NN + 7) / 8, 256, 0, stream>>>(qkvs, g2w, g2b, batch, gate, gmax);
    gexp_kernel<<<(NN + 255) / 256, 256, 0, stream>>>(gate, batch, gmax, gex, gden);
    gsum_kernel<<<(NN * 128) / 256, 256, 0, stream>>>(h, gex, gden, batch, gsum);
    readout_kernel<<<G_, 128, 0, stream>>>(gsum, r1w, r1b, r2w, r2b, (float*)d_out);
}
